// yolowrapper_89756226552376
// MI455X (gfx1250) — matrix operand order repaired and checked by reading, not compiled, not run on hardware
//
#include <hip/hip_runtime.h>
#include <hip/hip_bf16.h>

// ---------------- problem constants (from reference) ----------------
#define BB 32
#define AA 8400
#define NCC 80
#define KK 1024
#define GG 32
#define NSUPP 16
#define MAX_DETC 300
#define CONF_NMS_C 0.25f
#define IOU_NMS_C 0.45f
#define CONF_MATCH_C 0.5f
#define IOU_MATCH_C 0.6f
#define MAX_WH_C 7680.0f

typedef __attribute__((ext_vector_type(16))) _Float16 v16h;
typedef __attribute__((ext_vector_type(8)))  float    v8f;
typedef __attribute__((ext_vector_type(4)))  unsigned int v4u;
typedef __attribute__((ext_vector_type(8)))  int      v8i_t;
typedef __attribute__((ext_vector_type(4)))  int      v4i_t;

__device__ __forceinline__ float iou_xyxy(float ax1, float ay1, float ax2, float ay2,
                                          float bx1, float by1, float bx2, float by2) {
    float lx = fmaxf(ax1, bx1), ly = fmaxf(ay1, by1);
    float rx = fminf(ax2, bx2), ry = fminf(ay2, by2);
    float w  = fmaxf(rx - lx, 0.0f), h = fmaxf(ry - ly, 0.0f);
    float inter = w * h;
    float aa = (ax2 - ax1) * (ay2 - ay1);
    float ab = (bx2 - bx1) * (by2 - by1);
    return inter / (aa + ab - inter + 1e-9f);
}

// ---------------- kernel 1: decode + class max/argmax ----------------
__global__ void decode_kernel(const float* __restrict__ preds,
                              float* __restrict__ boxes_all,
                              float* __restrict__ conf_all,
                              int*   __restrict__ cls_all) {
    int idx = blockIdx.x * blockDim.x + threadIdx.x;
    if (idx >= BB * AA) return;
    int b = idx / AA, a = idx - b * AA;
    size_t base = (size_t)b * 84 * AA + a;

    float cx = preds[base + 0 * (size_t)AA];
    float cy = preds[base + 1 * (size_t)AA];
    float w  = preds[base + 2 * (size_t)AA];
    float h  = preds[base + 3 * (size_t)AA];

    float m = -1.0f;
    int   c = 0;
    for (int ci = 0; ci < NCC; ++ci) {
        const float* p = preds + base + (size_t)(4 + ci) * AA;
        if (ci + 4 < NCC) __builtin_prefetch(p + 4 * (size_t)AA, 0, 1);  // global_prefetch_b8
        float s = *p;
        if (s > m) { m = s; c = ci; }
    }
    float conf = (m > CONF_NMS_C) ? m : 0.0f;

    float hw = 0.5f * w, hh = 0.5f * h;
    size_t o = (size_t)idx * 4;
    boxes_all[o + 0] = cx - hw;
    boxes_all[o + 1] = cy - hh;
    boxes_all[o + 2] = cx + hw;
    boxes_all[o + 3] = cy + hh;
    conf_all[idx] = conf;
    cls_all[idx]  = c;
}

// ---------------- kernel 2: per-batch exact top-K via bitonic sort ----------------
#define SORT_N 16384
__global__ __launch_bounds__(1024)
void topk_kernel(const float* __restrict__ conf_all,
                 const int*   __restrict__ cls_all,
                 const float* __restrict__ boxes_all,
                 float* __restrict__ conf_k,
                 int*   __restrict__ cls_k,
                 float* __restrict__ boxes_k) {
    __shared__ unsigned long long skey[SORT_N];   // 128 KB (fits 320 KB WGP LDS)
    int b = blockIdx.x, tid = threadIdx.x;

    for (int i = tid; i < SORT_N; i += 1024) {
        unsigned long long kv = 0ull;
        if (i < AA) {
            float cf = conf_all[(size_t)b * AA + i];
            unsigned cb = __float_as_uint(cf);               // conf >= 0 -> monotonic bits
            kv = ((unsigned long long)cb << 32) | (unsigned long long)(0xFFFFFFFFu - (unsigned)i);
        }
        skey[i] = kv;
    }
    __syncthreads();

    // bitonic sort, descending
    for (unsigned k = 2; k <= SORT_N; k <<= 1) {
        for (unsigned j = k >> 1; j > 0; j >>= 1) {
            for (unsigned i = tid; i < SORT_N; i += 1024) {
                unsigned ixj = i ^ j;
                if (ixj > i) {
                    bool desc = ((i & k) == 0);
                    unsigned long long x = skey[i], y = skey[ixj];
                    bool sw = desc ? (x < y) : (x > y);
                    if (sw) { skey[i] = y; skey[ixj] = x; }
                }
            }
            __syncthreads();
        }
    }

    if (tid < KK) {
        unsigned long long kv = skey[tid];
        float cf = __uint_as_float((unsigned)(kv >> 32));
        unsigned idx = 0xFFFFFFFFu - (unsigned)(kv & 0xFFFFFFFFu);
        if (idx >= AA) idx = 0;                                 // padding entries (conf==0)
        size_t src = (size_t)b * AA + idx;
        size_t dst = (size_t)b * KK + tid;
        conf_k[dst] = cf;
        cls_k[dst]  = cls_all[src];
        boxes_k[dst * 4 + 0] = boxes_all[src * 4 + 0];
        boxes_k[dst * 4 + 1] = boxes_all[src * 4 + 1];
        boxes_k[dst * 4 + 2] = boxes_all[src * 4 + 2];
        boxes_k[dst * 4 + 3] = boxes_all[src * 4 + 3];
    }
}

// ---------------- kernel 3: NMS (bit-matrix in LDS + wave32 sequential scan) ----------------
__global__ __launch_bounds__(1024)
void nms_kernel(const float* __restrict__ boxes_k,
                const float* __restrict__ conf_k,
                const int*   __restrict__ cls_k,
                int* __restrict__ keep_f) {
    __shared__ float4 bs[KK];                    // class-shifted boxes, 16 KB
    __shared__ float  cs[KK];                    // conf, 4 KB
    __shared__ unsigned mat[KK][32];             // suppression bit-matrix, 128 KB
    int b = blockIdx.x, tid = threadIdx.x;
    size_t bo = (size_t)b * KK;

    float shift = (float)cls_k[bo + tid] * MAX_WH_C;
    float4 bb;
    bb.x = boxes_k[(bo + tid) * 4 + 0] + shift;
    bb.y = boxes_k[(bo + tid) * 4 + 1] + shift;
    bb.z = boxes_k[(bo + tid) * 4 + 2] + shift;
    bb.w = boxes_k[(bo + tid) * 4 + 3] + shift;
    bs[tid] = bb;
    cs[tid] = conf_k[bo + tid];
    __syncthreads();

    // row tid of the (iou > thresh) bit matrix
    float4 a = bs[tid];
    for (int w = 0; w < 32; ++w) {
        unsigned bits = 0u;
        for (int l = 0; l < 32; ++l) {
            float4 o = bs[w * 32 + l];
            float v = iou_xyxy(a.x, a.y, a.z, a.w, o.x, o.y, o.z, o.w);
            bits |= (v > IOU_NMS_C) ? (1u << l) : 0u;
        }
        mat[tid][w] = bits;
    }
    __syncthreads();

    // sequential greedy scan on wave 0 (lane owns one 32-bit word of the keep mask)
    if (tid < 32) {
        int lane = tid;
        unsigned keepw = 0u;
        int cnt = 0;
        #pragma unroll 1
        for (int i = 0; i < KK; ++i) {
            unsigned myw = keepw & mat[i][lane];
            bool sup = (__ballot(myw != 0u) != 0ull);          // uniform
            bool valid = (cs[i] > 0.0f);                        // uniform
            bool k = valid && !sup;
            if (k) {
                if ((i >> 5) == lane) keepw |= 1u << (i & 31);
                cnt++;                                          // cumsum including i
            }
            if (lane == 0) keep_f[bo + i] = (k && cnt <= MAX_DETC) ? 1 : 0;
        }
    }
}

// ---------------- kernel 4: matching + WMMA one-hot histograms ----------------
__global__ __launch_bounds__(256)
void match_kernel(const float* __restrict__ boxes_k,
                  const float* __restrict__ conf_k,
                  const int*   __restrict__ cls_k,
                  const int*   __restrict__ keep_f,
                  const float* __restrict__ gt_boxes,
                  const int*   __restrict__ gt_cls,
                  const int*   __restrict__ gt_mask,
                  const int*   __restrict__ cat_to_super,
                  float* __restrict__ out) {
    __shared__ float sh_gt[GG * 4];              // TDM destination (512 B)
    __shared__ int   sh_cls[GG];
    __shared__ float sh_msk[GG];
    __shared__ int   sh_c2s[NCC];
    __shared__ float red_iou[GG][8];
    __shared__ int   red_idx[GG][8];
    __shared__ float gbest_iou[GG], gbest_conf[GG];
    __shared__ int   gn_pred;
    __shared__ float sh_cat[NCC], sh_sup[NSUPP];

    int b = blockIdx.x, tid = threadIdx.x;
    size_t bo = (size_t)b * KK;

    if (tid < 32) {
        // DMA the 32x4 f32 GT tile into LDS with the Tensor Data Mover.
#if __has_builtin(__builtin_amdgcn_tensor_load_to_lds)
        unsigned int lds_off = (unsigned int)(unsigned long long)(void*)&sh_gt[0];
        unsigned long long ga = (unsigned long long)(const void*)(gt_boxes + (size_t)b * GG * 4);
        v4u g0;
        g0[0] = 1u;                                            // count=1 valid descriptor
        g0[1] = lds_off;                                       // lds_addr
        g0[2] = (unsigned)(ga & 0xFFFFFFFFu);                  // global_addr[31:0]
        g0[3] = (unsigned)((ga >> 32) & 0x01FFFFFFu) | (2u << 30); // addr[56:32] | type=2
        v8i_t g1;
        g1[0] = 0x20000;                  // data_size=4B (code 2), no multicast/pad
        g1[1] = (int)(128u << 16);        // tensor_dim0 = 128 elements (lo16 in [31:16])
        g1[2] = (int)(1u << 16);          // tensor_dim0 hi=0 | tensor_dim1 = 1
        g1[3] = (int)(128u << 16);        // tile_dim0 = 128
        g1[4] = 1;                        // tile_dim1 = 1, tile_dim2 = 0
        g1[5] = 128;                      // tensor_dim0_stride = 128
        g1[6] = 0;
        g1[7] = 0;
        v4i_t gz = {};
#if __clang_major__ >= 23
        v8i_t gx = {};
        __builtin_amdgcn_tensor_load_to_lds(g0, g1, gz, gz, gx, 0);
#else
        __builtin_amdgcn_tensor_load_to_lds(g0, g1, gz, gz, 0);
#endif
        __builtin_amdgcn_s_wait_tensorcnt(0);
#else
        for (int i = tid; i < GG * 4; i += 32)
            sh_gt[i] = gt_boxes[(size_t)b * GG * 4 + i];
#endif
    } else if (tid < 64) {
        sh_cls[tid - 32] = gt_cls[(size_t)b * GG + (tid - 32)];
    } else if (tid < 96) {
        sh_msk[tid - 64] = (gt_mask[(size_t)b * GG + (tid - 64)] != 0) ? 1.0f : 0.0f;
    } else if (tid < 96 + NCC) {
        sh_c2s[tid - 96] = cat_to_super[tid - 96];
    }
    if (tid == 200) gn_pred = 0;
    __syncthreads();

    // n_pred = sum over preds of keep & (conf > CONF_MATCH)
    {
        int base = tid * 4, local = 0;
        for (int i = base; i < base + 4; ++i)
            if (keep_f[bo + i] && conf_k[bo + i] > CONF_MATCH_C) local++;
        if (local) atomicAdd(&gn_pred, local);
    }

    // per-(gt, chunk) best IoU / first-index argmax over eligible preds
    {
        int g = tid & 31, ch = tid >> 5;
        float gx1 = sh_gt[g * 4 + 0], gy1 = sh_gt[g * 4 + 1];
        float gx2 = sh_gt[g * 4 + 2], gy2 = sh_gt[g * 4 + 3];
        int gc = sh_cls[g];
        float bi = -1.0f; int bidx = 0;
        for (int i = ch * 128; i < ch * 128 + 128; ++i) {
            bool pv = keep_f[bo + i] && (conf_k[bo + i] > CONF_MATCH_C);
            bool elig = pv && (cls_k[bo + i] == gc);
            float v = -1.0f;
            if (elig) {
                v = iou_xyxy(boxes_k[(bo + i) * 4 + 0], boxes_k[(bo + i) * 4 + 1],
                             boxes_k[(bo + i) * 4 + 2], boxes_k[(bo + i) * 4 + 3],
                             gx1, gy1, gx2, gy2);
            }
            if (v > bi) { bi = v; bidx = i; }
        }
        red_iou[g][ch] = bi; red_idx[g][ch] = bidx;
    }
    __syncthreads();

    if (tid < 32) {
        int g = tid;
        float bi = red_iou[g][0]; int bidx = red_idx[g][0];
        for (int c = 1; c < 8; ++c)
            if (red_iou[g][c] > bi) { bi = red_iou[g][c]; bidx = red_idx[g][c]; }
        gbest_iou[g]  = bi;
        gbest_conf[g] = conf_k[bo + bidx];
    }
    __syncthreads();

    // cat/super histograms as one-hot matmuls on the matrix core:
    // counts(1x80) = mask(1x32) x onehot(32x80), five 16-col tiles + one super tile.
    if (tid < 32) {
        int lane = tid;
        v16h aV;
        #pragma unroll
        for (int h = 0; h < 16; ++h) {
            int kk = h + ((lane >> 4) << 4);                     // A 16x32 f16 layout (contiguous k = 16*half + h)
            aV[h] = (_Float16)sh_msk[kk];
        }
        int n = lane & 15;
        int kbase = (lane >> 4) << 4;                            // B 32x16 f16 layout
        float catv[5];
        #pragma unroll
        for (int t = 0; t < 5; ++t) {
            v16h bV;
            #pragma unroll
            for (int h = 0; h < 16; ++h)
                bV[h] = (_Float16)((sh_cls[kbase + h] == (t * 16 + n)) ? 1.0f : 0.0f);
            v8f cz = {};
            v8f d = __builtin_amdgcn_wmma_f32_16x16x32_f16(false, aV, false, bV,
                                                           (short)0, cz, false, false);
            catv[t] = d[0];
        }
        v16h bS;
        #pragma unroll
        for (int h = 0; h < 16; ++h)
            bS[h] = (_Float16)((sh_c2s[sh_cls[kbase + h]] == n) ? 1.0f : 0.0f);
        v8f cz2 = {};
        v8f dS = __builtin_amdgcn_wmma_f32_16x16x32_f16(false, aV, false, bS,
                                                        (short)0, cz2, false, false);
        if (lane < 16) {
            #pragma unroll
            for (int t = 0; t < 5; ++t) sh_cat[t * 16 + lane] = catv[t];
            sh_sup[lane] = dS[0];
        }
    }
    __syncthreads();

    if (tid == 0) {
        float n_gt = 0.f, n_hit = 0.f, sum_iou = 0.f, sum_conf = 0.f;
        for (int g = 0; g < GG; ++g) {
            float m = sh_msk[g];
            n_gt += m;
            bool hit = (gbest_iou[g] > IOU_MATCH_C) && (m > 0.0f);
            if (hit) { n_hit += 1.0f; sum_iou += gbest_iou[g]; sum_conf += gbest_conf[g]; }
        }
        float bm = sh_cat[0]; int mfc = 0;
        for (int c = 1; c < NCC; ++c) if (sh_cat[c] > bm) { bm = sh_cat[c]; mfc = c; }
        float bsv = sh_sup[0]; int mfs = 0;
        for (int s = 1; s < NSUPP; ++s) if (sh_sup[s] > bsv) { bsv = sh_sup[s]; mfs = s; }

        float fh = n_hit;
        float mean_iou  = sum_iou / fmaxf(fh, 1.0f);
        float mean_conf = (n_hit > 0.0f) ? (sum_conf / fmaxf(fh, 1.0f)) : 1.0f;
        float correct   = fh / fmaxf(n_gt, 1.0f);

        float r0, r1, r2, r3, r4;
        if (n_gt == 0.0f) {
            bool np0 = (gn_pred == 0);
            r0 = np0 ? 1.0f : 0.0f; r1 = 1.0f;
            r2 = np0 ? -1.0f : -2.0f; r3 = np0 ? -1.0f : -2.0f;
            r4 = np0 ? 1.0f : 0.0f;
        } else {
            r0 = mean_iou; r1 = mean_conf; r2 = (float)mfc; r3 = (float)mfs; r4 = correct;
        }
        float* o = out + (size_t)b * 5;
        o[0] = r0; o[1] = r1; o[2] = r2; o[3] = r3; o[4] = r4;
    }
}

// ---------------- host launcher ----------------
extern "C" void kernel_launch(void* const* d_in, const int* in_sizes, int n_in,
                              void* d_out, int out_size, void* d_ws, size_t ws_size,
                              hipStream_t stream) {
    const float* preds        = (const float*)d_in[0];
    const float* gt_boxes     = (const float*)d_in[1];
    const int*   gt_cls       = (const int*)d_in[2];
    const int*   gt_mask      = (const int*)d_in[3];
    const int*   cat_to_super = (const int*)d_in[4];
    float* out = (float*)d_out;

    char* w = (char*)d_ws;
    float* boxes_all = (float*)w; w += (size_t)BB * AA * 4 * sizeof(float);
    float* conf_all  = (float*)w; w += (size_t)BB * AA * sizeof(float);
    int*   cls_all   = (int*)w;   w += (size_t)BB * AA * sizeof(int);
    float* conf_k    = (float*)w; w += (size_t)BB * KK * sizeof(float);
    float* boxes_k   = (float*)w; w += (size_t)BB * KK * 4 * sizeof(float);
    int*   cls_k     = (int*)w;   w += (size_t)BB * KK * sizeof(int);
    int*   keep_f    = (int*)w;   w += (size_t)BB * KK * sizeof(int);
    (void)ws_size; (void)in_sizes; (void)n_in; (void)out_size;

    int n = BB * AA;
    decode_kernel<<<(n + 255) / 256, 256, 0, stream>>>(preds, boxes_all, conf_all, cls_all);
    topk_kernel<<<BB, 1024, 0, stream>>>(conf_all, cls_all, boxes_all, conf_k, cls_k, boxes_k);
    nms_kernel<<<BB, 1024, 0, stream>>>(boxes_k, conf_k, cls_k, keep_f);
    match_kernel<<<BB, 256, 0, stream>>>(boxes_k, conf_k, cls_k, keep_f,
                                         gt_boxes, gt_cls, gt_mask, cat_to_super, out);
}
